// ParticleNet_11218454577374
// MI455X (gfx1250) — compile-verified, matrix operand order checked by reading, not run on hardware
//
#include <hip/hip_runtime.h>
#include <hip/hip_bf16.h>

// ---------------------------------------------------------------------------
// ParticleNet forward for MI455X (gfx1250, wave32, WMMA).
//
//   0) input BN stats -> finalize -> apply                    (VALU)
//   1) EdgeConv1: 3x [WMMA GEMM + BN-stats] + shortcut GEMM + combine
//   2) kNN rebuild on 32-dim features                         (VALU)
//   3) EdgeConv2: same with C=64
//   4) node mean-pool + FC head                               (VALU)
//
// GEMM kernels: v_wmma_f32_16x16x32_f16. B fragments are built once in LDS,
// then hoisted into registers per wave (loop-invariant weights); A fragments
// come from aligned b128 loads with the edge gather / previous layer's
// BN+ReLU fused in; BN statistics accumulate in registers across all tiles
// and reduce once per wave; D tiles staged in LDS and written as b128 row
// chunks. Inter-layer edge activations kept f16: the problem is bound by
// ~1.4 GB of activation traffic at 23.3 TB/s, not by compute.
// ---------------------------------------------------------------------------

typedef __attribute__((ext_vector_type(16))) _Float16 v16h;
typedef __attribute__((ext_vector_type(8)))  float    v8f;

#define BATCH   2048
#define NNODE   128
#define KNBR    7
#define NUMN    (BATCH * NNODE)          // 262144 nodes
#define MEDGE   (NUMN * KNBR)            // 1835008 edges (divisible by 16)
#define EPS_BN  1e-5f

// K-offset within a 32-wide chunk for the 16-bit WMMA A/B fragment layout:
// lanes 0-15 hold K {0..7,16..23}, lanes 16-31 hold K {8..15,24..31};
// fragment element j in [0,8) -> K = base + j, j in [8,16) -> K = base+16+(j-8).
__device__ __forceinline__ int frag_koff(int lane, int j) {
    return ((lane & 16) ? 8 : 0) + ((j & 8) ? 16 : 0) + (j & 7);
}

__device__ __forceinline__ void wave_lds_fence() {
    __builtin_amdgcn_wave_barrier();
    asm volatile("s_wait_dscnt 0x0" ::: "memory");
}

__device__ __forceinline__ void set8(v16h& v, int base, const float4& a, const float4& b) {
    v[base + 0] = (_Float16)a.x; v[base + 1] = (_Float16)a.y;
    v[base + 2] = (_Float16)a.z; v[base + 3] = (_Float16)a.w;
    v[base + 4] = (_Float16)b.x; v[base + 5] = (_Float16)b.y;
    v[base + 6] = (_Float16)b.z; v[base + 7] = (_Float16)b.w;
}

// ---------------------------------------------------------------------------
// GEMM + BN-statistics kernel.
//   MODE 0: A row = concat(x_i, x_j - x_i) gathered from fts (f32, CIN ch)
//   MODE 1: A row = relu(z * scaleA + shiftA), z f16 with row stride KCHUNKS*32
//   MODE 2: A row = fts row (f32, CIN channels)               [shortcut]
// Cout = NTILES*16 (compile time). Writes z = A@W + bias (f16 [M, Cout]) and
// accumulates per-channel sum/sumsq into stats[0..Cout)/stats[Cout..2Cout).
// ---------------------------------------------------------------------------
template <int MODE, int CIN, int KCHUNKS, int NTILES>
__global__ __launch_bounds__(256)
void gemm_bnstats(const float* __restrict__ fts,
                  const int* __restrict__ idx,
                  const _Float16* __restrict__ zin,
                  const float* __restrict__ scaleA,
                  const float* __restrict__ shiftA,
                  const float* __restrict__ W,     // [Kdim, Cout] row-major f32
                  const float* __restrict__ bias,  // [Cout]
                  int Kdim, int M,
                  _Float16* __restrict__ zout,
                  float* __restrict__ stats)
{
    constexpr int COUT = NTILES * 16;

    __shared__ __align__(32) _Float16 sB[KCHUNKS * NTILES * 32 * 16];
    __shared__ __align__(16) _Float16 sD[8 * 16 * COUT];
    __shared__ float s_scale[64];
    __shared__ float s_shift[64];
    __shared__ float s_sum[64];
    __shared__ float s_sq[64];

    const int tid = threadIdx.x;
    if (tid < COUT) { s_sum[tid] = 0.f; s_sq[tid] = 0.f; }
    if (MODE == 1 && tid < KCHUNKS * 32) {
        s_scale[tid] = scaleA[tid];
        s_shift[tid] = shiftA[tid];
    }
    // Build all B fragments (WMMA layout) in LDS: one slot per (kc,nt,lane).
    for (int s = tid; s < KCHUNKS * NTILES * 32; s += 256) {
        const int plane = s & 31;
        const int pnt   = (s >> 5) & (NTILES - 1);
        const int pkc   = s / (32 * NTILES);
        const int col   = pnt * 16 + (plane & 15);
#pragma unroll
        for (int j = 0; j < 16; ++j) {
            const int K = pkc * 32 + frag_koff(plane, j);
            sB[s * 16 + j] = (K < Kdim) ? (_Float16)W[(size_t)K * COUT + col]
                                        : (_Float16)0.f;
        }
    }
    __syncthreads();

    const int lane    = tid & 31;
    const int lane16  = lane & 15;
    const int halfsel = lane >> 4;
    const int wv      = tid >> 5;
    const int wid     = (blockIdx.x * blockDim.x + tid) >> 5;
    const int nwaves  = (gridDim.x * blockDim.x) >> 5;
    const int ntiles  = M >> 4;
    _Float16* sDw     = &sD[wv * 16 * COUT];

    // ---- hoist loop-invariant B fragments and bias into registers ----
    v16h  bf[KCHUNKS][NTILES];
    float bcol[NTILES];
#pragma unroll
    for (int kc = 0; kc < KCHUNKS; ++kc)
#pragma unroll
        for (int nt = 0; nt < NTILES; ++nt)
            bf[kc][nt] = *reinterpret_cast<const v16h*>(
                &sB[((kc * NTILES + nt) * 32 + lane) * 16]);
#pragma unroll
    for (int nt = 0; nt < NTILES; ++nt) bcol[nt] = bias[nt * 16 + lane16];

    // per-lane running BN statistics (per output column)
    float accsum[NTILES];
    float accsq[NTILES];
#pragma unroll
    for (int nt = 0; nt < NTILES; ++nt) { accsum[nt] = 0.f; accsq[nt] = 0.f; }

    for (int t = wid; t < ntiles; t += nwaves) {
        const int row = t * 16 + lane16;

        // ---- A fragments ----
        v16h af[KCHUNKS];
        if (MODE == 0 && CIN == 5) {
            const int node   = row / KNBR;
            const int bgraph = node >> 7;
            const int jn     = idx[row];
            const float* xi  = fts + (size_t)node * 5;
            const float* xj  = fts + ((size_t)(bgraph << 7) + jn) * 5;
            float xv[5], dv[5];
#pragma unroll
            for (int c = 0; c < 5; ++c) { xv[c] = xi[c]; dv[c] = xj[c] - xv[c]; }
            v16h a;
#pragma unroll
            for (int j = 0; j < 16; ++j) a[j] = (_Float16)0.f;
            if (halfsel == 0) {            // K = 0..7 (j0-7), 16..23 (j8-15, all 0)
                a[0] = (_Float16)xv[0]; a[1] = (_Float16)xv[1]; a[2] = (_Float16)xv[2];
                a[3] = (_Float16)xv[3]; a[4] = (_Float16)xv[4];
                a[5] = (_Float16)dv[0]; a[6] = (_Float16)dv[1]; a[7] = (_Float16)dv[2];
            } else {                       // K = 8..15 (j0-7): only 8,9 live
                a[0] = (_Float16)dv[3]; a[1] = (_Float16)dv[4];
            }
            af[0] = a;
        } else if (MODE == 0) {            // CIN == 32, KCHUNKS == 2
            const int node   = row / KNBR;
            const int bgraph = node >> 7;
            const int jn     = idx[row];
            const float* xi  = fts + (size_t)node * 32;
            const float* xj  = fts + ((size_t)(bgraph << 7) + jn) * 32;
#pragma unroll
            for (int g = 0; g < 2; ++g) {
                const int off = g * 16 + (halfsel ? 8 : 0);
                const float4 xa = *reinterpret_cast<const float4*>(xi + off);
                const float4 xb = *reinterpret_cast<const float4*>(xi + off + 4);
                const float4 ya = *reinterpret_cast<const float4*>(xj + off);
                const float4 yb = *reinterpret_cast<const float4*>(xj + off + 4);
                float4 da, db;
                da.x = ya.x - xa.x; da.y = ya.y - xa.y; da.z = ya.z - xa.z; da.w = ya.w - xa.w;
                db.x = yb.x - xb.x; db.y = yb.y - xb.y; db.z = yb.z - xb.z; db.w = yb.w - xb.w;
                set8(af[0], g * 8, xa, xb);   // K in [0,32): x_i
                set8(af[1], g * 8, da, db);   // K in [32,64): x_j - x_i
            }
        } else if (MODE == 1) {
            constexpr int STRIDE = KCHUNKS * 32;
            const _Float16* zr = zin + (size_t)row * STRIDE;
#pragma unroll
            for (int kc = 0; kc < KCHUNKS; ++kc) {
#pragma unroll
                for (int g = 0; g < 2; ++g) {
                    const int off = kc * 32 + g * 16 + (halfsel ? 8 : 0);
                    const float4 raw = *reinterpret_cast<const float4*>(zr + off);
                    _Float16 h8[8];
                    __builtin_memcpy(h8, &raw, 16);
#pragma unroll
                    for (int jj = 0; jj < 8; ++jj) {
                        const int K = off + jj;
                        const float v = fmaxf(fmaf((float)h8[jj], s_scale[K], s_shift[K]), 0.f);
                        af[kc][g * 8 + jj] = (_Float16)v;
                    }
                }
            }
        } else if (MODE == 2 && CIN == 5) {
            const float* xr = fts + (size_t)row * 5;
            v16h a;
#pragma unroll
            for (int j = 0; j < 16; ++j) a[j] = (_Float16)0.f;
            if (halfsel == 0) {
                a[0] = (_Float16)xr[0]; a[1] = (_Float16)xr[1]; a[2] = (_Float16)xr[2];
                a[3] = (_Float16)xr[3]; a[4] = (_Float16)xr[4];
            }
            af[0] = a;
        } else {                           // MODE 2, CIN == 32
            const float* xr = fts + (size_t)row * 32;
#pragma unroll
            for (int g = 0; g < 2; ++g) {
                const int off = g * 16 + (halfsel ? 8 : 0);
                const float4 xa = *reinterpret_cast<const float4*>(xr + off);
                const float4 xb = *reinterpret_cast<const float4*>(xr + off + 4);
                set8(af[0], g * 8, xa, xb);
            }
        }

        // ---- output column tiles: WMMAs straight from registers ----
#pragma unroll
        for (int nt = 0; nt < NTILES; ++nt) {
            v8f acc;
#pragma unroll
            for (int d = 0; d < 8; ++d) acc[d] = 0.f;
#pragma unroll
            for (int kc = 0; kc < KCHUNKS; ++kc)
                acc = __builtin_amdgcn_wmma_f32_16x16x32_f16(
                          false, af[kc], false, bf[kc][nt], (short)0, acc, false, false);
#pragma unroll
            for (int d = 0; d < 8; ++d) {
                const float vv = acc[d] + bcol[nt];
                sDw[(d + halfsel * 8) * COUT + nt * 16 + lane16] = (_Float16)vv;
                accsum[nt] += vv;
                accsq[nt]  += vv * vv;
            }
        }

        // ---- coalesced b128 write-out of the 16 x COUT f16 tile ----
        wave_lds_fence();
#pragma unroll
        for (int c = lane; c < 2 * COUT; c += 32) {      // 8-half chunks
            const int r  = c / (COUT / 8);
            const int cc = c % (COUT / 8);
            const float4 v = *reinterpret_cast<const float4*>(&sDw[r * COUT + cc * 8]);
            *reinterpret_cast<float4*>(&zout[(size_t)(t * 16 + r) * COUT + cc * 8]) = v;
        }
        wave_lds_fence();
    }

    // ---- one stats reduction per wave ----
#pragma unroll
    for (int nt = 0; nt < NTILES; ++nt) {
        float psum = accsum[nt] + __shfl_xor(accsum[nt], 16);
        float psq  = accsq[nt]  + __shfl_xor(accsq[nt], 16);
        if (halfsel == 0) {
            atomicAdd(&s_sum[nt * 16 + lane16], psum);
            atomicAdd(&s_sq[nt * 16 + lane16],  psq);
        }
    }

    __syncthreads();
    if (tid < COUT) {
        atomicAdd(&stats[tid],        s_sum[tid]);
        atomicAdd(&stats[COUT + tid], s_sq[tid]);
    }
}

// ---------------------------------------------------------------------------
// BN finalize: fold mean/var/gamma/beta into per-channel scale/shift.
// ---------------------------------------------------------------------------
__global__ void bn_finalize(const float* __restrict__ stats,
                            const float* __restrict__ gamma,
                            const float* __restrict__ beta,
                            float invM, int C,
                            float* __restrict__ scale,
                            float* __restrict__ shift)
{
    const int c = blockIdx.x * blockDim.x + threadIdx.x;
    if (c >= C) return;
    const float m   = stats[c] * invM;
    const float var = stats[C + c] * invM - m * m;
    const float s   = gamma[c] * rsqrtf(var + EPS_BN);
    scale[c] = s;
    shift[c] = beta[c] - m * s;
}

// ---------------------------------------------------------------------------
// Per-channel sum/sumsq over the raw input features (C = 5).
// ---------------------------------------------------------------------------
__global__ void reduce_input(const float* __restrict__ x, size_t n,
                             float* __restrict__ stats)
{
    __shared__ float ss[5];
    __shared__ float sq[5];
    if (threadIdx.x < 5) { ss[threadIdx.x] = 0.f; sq[threadIdx.x] = 0.f; }
    __syncthreads();

    float ls[5] = {0.f, 0.f, 0.f, 0.f, 0.f};
    float lq[5] = {0.f, 0.f, 0.f, 0.f, 0.f};
    const size_t stride = (size_t)gridDim.x * blockDim.x;
    for (size_t i = blockIdx.x * blockDim.x + threadIdx.x; i < n; i += stride) {
        const float v = x[i];
        const int   c = (int)(i % 5);
        ls[c] += v;
        lq[c] += v * v;
    }
#pragma unroll
    for (int c = 0; c < 5; ++c) {
        atomicAdd(&ss[c], ls[c]);
        atomicAdd(&sq[c], lq[c]);
    }
    __syncthreads();
    if (threadIdx.x < 5) {
        atomicAdd(&stats[threadIdx.x],     ss[threadIdx.x]);
        atomicAdd(&stats[5 + threadIdx.x], sq[threadIdx.x]);
    }
}

__global__ void bn_apply(const float* __restrict__ x,
                         const float* __restrict__ scale,
                         const float* __restrict__ shift,
                         float* __restrict__ out, size_t n, int C)
{
    const size_t i = (size_t)blockIdx.x * blockDim.x + threadIdx.x;
    if (i >= n) return;
    const int c = (int)(i % C);
    out[i] = fmaf(x[i], scale[c], shift[c]);
}

// ---------------------------------------------------------------------------
// Combine: fts_out = relu( mean_k relu(bn(h3)) + bn(z_sc) )
// ---------------------------------------------------------------------------
__global__ void combine_edges(const _Float16* __restrict__ h3,
                              const float* __restrict__ sA,
                              const float* __restrict__ shA,
                              const _Float16* __restrict__ zsc,
                              const float* __restrict__ sS,
                              const float* __restrict__ shS,
                              float* __restrict__ ftsOut, int C)
{
    const size_t i = (size_t)blockIdx.x * blockDim.x + threadIdx.x;
    const size_t n = (size_t)NUMN * C;
    if (i >= n) return;
    const size_t node = i / C;
    const int    c    = (int)(i % C);
    const _Float16* hp = h3 + (node * KNBR) * C + c;
    float acc = 0.f;
#pragma unroll
    for (int kk = 0; kk < KNBR; ++kk)
        acc += fmaxf(fmaf((float)hp[(size_t)kk * C], sA[c], shA[c]), 0.f);
    acc *= (1.f / (float)KNBR);
    const float sc = fmaf((float)zsc[node * C + c], sS[c], shS[c]);
    ftsOut[i] = fmaxf(acc + sc, 0.f);
}

// ---------------------------------------------------------------------------
// kNN (k = 7, self excluded) on 32-dim features; one block per graph.
// ---------------------------------------------------------------------------
__global__ __launch_bounds__(128)
void knn_kernel(const float* __restrict__ fts, int* __restrict__ idxOut)
{
    __shared__ float xs[NNODE * 32];
    __shared__ float sq[NNODE];
    const int b = blockIdx.x;
    const int i = threadIdx.x;
    const float* src = fts + ((size_t)b * NNODE + i) * 32;
    float s = 0.f;
#pragma unroll
    for (int c = 0; c < 32; ++c) { const float v = src[c]; xs[i * 32 + c] = v; s += v * v; }
    sq[i] = s;
    __syncthreads();

    float bestd[KNBR];
    int   bestj[KNBR];
#pragma unroll
    for (int t = 0; t < KNBR; ++t) { bestd[t] = 3.0e38f; bestj[t] = 0; }

    for (int j = 0; j < NNODE; ++j) {
        if (j == i) continue;
        float dot = 0.f;
#pragma unroll
        for (int c = 0; c < 32; ++c) dot += xs[i * 32 + c] * xs[j * 32 + c];
        const float d = sq[i] + sq[j] - 2.f * dot;
        if (d < bestd[KNBR - 1]) {
            bestd[KNBR - 1] = d;
            bestj[KNBR - 1] = j;
#pragma unroll
            for (int t = KNBR - 1; t > 0; --t) {
                if (bestd[t] < bestd[t - 1]) {
                    const float td = bestd[t]; bestd[t] = bestd[t - 1]; bestd[t - 1] = td;
                    const int   tj = bestj[t]; bestj[t] = bestj[t - 1]; bestj[t - 1] = tj;
                }
            }
        }
    }
#pragma unroll
    for (int t = 0; t < KNBR; ++t)
        idxOut[((size_t)b * NNODE + i) * KNBR + t] = bestj[t];
}

// ---------------------------------------------------------------------------
// Mean-pool over nodes + FC head (64 -> 128 relu -> 2); one block per graph.
// ---------------------------------------------------------------------------
__global__ __launch_bounds__(128)
void head_kernel(const float* __restrict__ fts,   // [NUMN, 64]
                 const float* __restrict__ W1,    // [64,128]
                 const float* __restrict__ b1,    // [128]
                 const float* __restrict__ W2,    // [128,2]
                 const float* __restrict__ b2,    // [2]
                 float* __restrict__ out)         // [BATCH,2]
{
    __shared__ float mean[64];
    __shared__ float h[128];
    const int b = blockIdx.x;
    const int t = threadIdx.x;
    if (t < 64) {
        const float* p = fts + (size_t)b * NNODE * 64 + t;
        float s = 0.f;
        for (int n = 0; n < NNODE; ++n) s += p[(size_t)n * 64];
        mean[t] = s * (1.f / (float)NNODE);
    }
    __syncthreads();
    float acc = b1[t];
    for (int c = 0; c < 64; ++c) acc = fmaf(mean[c], W1[c * 128 + t], acc);
    h[t] = fmaxf(acc, 0.f);
    __syncthreads();
    if (t < 2) {
        float o = b2[t];
        for (int u = 0; u < 128; ++u) o = fmaf(h[u], W2[u * 2 + t], o);
        out[(size_t)b * 2 + t] = o;
    }
}

// ---------------------------------------------------------------------------
// Host launcher.
// Input order (JAX pytree flatten, dict keys sorted):
//  0 features  1 knn_idx
//  2 bn_fts.b  3 bn_fts.g
//  conv0: L0 {4 W,5 b,6 bt,7 g} L1 {8..11} L2 {12..15} sc {16 W,17 b,18 bt,19 g}
//  conv1: L0 {20..23} L1 {24..27} L2 {28..31} sc {32..35}
//  fc: 36 W1, 37 W2, 38 b1, 39 b2
// ---------------------------------------------------------------------------
extern "C" void kernel_launch(void* const* d_in, const int* in_sizes, int n_in,
                              void* d_out, int out_size, void* d_ws, size_t ws_size,
                              hipStream_t stream)
{
    (void)in_sizes; (void)n_in; (void)out_size; (void)ws_size;

    const float* features = (const float*)d_in[0];
    const int*   knn0     = (const int*)d_in[1];
    const float* P[40];
    for (int i = 2; i < 40; ++i) P[i] = (const float*)d_in[i];
    float* out = (float*)d_out;

    // ---- workspace carve-up ----
    char* w = (char*)d_ws;
    float*    fts   = (float*)w;            w += (size_t)NUMN * 64 * sizeof(float);
    _Float16* zA    = (_Float16*)w;         w += (size_t)MEDGE * 64 * sizeof(_Float16);
    _Float16* zB    = (_Float16*)w;         w += (size_t)MEDGE * 64 * sizeof(_Float16);
    _Float16* zsc   = (_Float16*)w;         w += (size_t)NUMN * 64 * sizeof(_Float16);
    int*      idx2  = (int*)w;              w += (size_t)NUMN * KNBR * sizeof(int);
    float*    stats = (float*)w;            // 9 groups x 256 floats

    float* G[9];
    for (int g = 0; g < 9; ++g) G[g] = stats + g * 256;
    hipMemsetAsync(stats, 0, 9 * 256 * sizeof(float), stream);

    const float invNodes = 1.f / (float)NUMN;
    const float invEdges = 1.f / (float)MEDGE;

    // ---- input BN ----
    reduce_input<<<512, 256, 0, stream>>>(features, (size_t)NUMN * 5, G[0]);
    bn_finalize<<<1, 64, 0, stream>>>(G[0], P[3], P[2], invNodes, 5, G[0] + 128, G[0] + 192);
    {
        const size_t n = (size_t)NUMN * 5;
        bn_apply<<<(int)((n + 255) / 256), 256, 0, stream>>>(features, G[0] + 128, G[0] + 192, fts, n, 5);
    }

    // ---- EdgeConv block 1 (Cin=5, Cout=32) ----
    gemm_bnstats<0, 5, 1, 2><<<896, 256, 0, stream>>>(fts, knn0, nullptr, nullptr, nullptr,
                                                      P[4], P[5], 10, MEDGE, zA, G[1]);
    bn_finalize<<<1, 64, 0, stream>>>(G[1], P[7], P[6], invEdges, 32, G[1] + 128, G[1] + 192);
    gemm_bnstats<1, 0, 1, 2><<<896, 256, 0, stream>>>(nullptr, nullptr, zA, G[1] + 128, G[1] + 192,
                                                      P[8], P[9], 32, MEDGE, zB, G[2]);
    bn_finalize<<<1, 64, 0, stream>>>(G[2], P[11], P[10], invEdges, 32, G[2] + 128, G[2] + 192);
    gemm_bnstats<1, 0, 1, 2><<<896, 256, 0, stream>>>(nullptr, nullptr, zB, G[2] + 128, G[2] + 192,
                                                      P[12], P[13], 32, MEDGE, zA, G[3]);
    bn_finalize<<<1, 64, 0, stream>>>(G[3], P[15], P[14], invEdges, 32, G[3] + 128, G[3] + 192);
    gemm_bnstats<2, 5, 1, 2><<<128, 256, 0, stream>>>(fts, nullptr, nullptr, nullptr, nullptr,
                                                      P[16], P[17], 5, NUMN, zsc, G[4]);
    bn_finalize<<<1, 64, 0, stream>>>(G[4], P[19], P[18], invNodes, 32, G[4] + 128, G[4] + 192);
    {
        const size_t n = (size_t)NUMN * 32;
        combine_edges<<<(int)((n + 255) / 256), 256, 0, stream>>>(
            zA, G[3] + 128, G[3] + 192, zsc, G[4] + 128, G[4] + 192, fts, 32);
    }

    // ---- kNN rebuild on 32-dim features ----
    knn_kernel<<<BATCH, 128, 0, stream>>>(fts, idx2);

    // ---- EdgeConv block 2 (Cin=32, Cout=64) ----
    gemm_bnstats<0, 32, 2, 4><<<896, 256, 0, stream>>>(fts, idx2, nullptr, nullptr, nullptr,
                                                       P[20], P[21], 64, MEDGE, zA, G[5]);
    bn_finalize<<<1, 64, 0, stream>>>(G[5], P[23], P[22], invEdges, 64, G[5] + 128, G[5] + 192);
    gemm_bnstats<1, 0, 2, 4><<<896, 256, 0, stream>>>(nullptr, nullptr, zA, G[5] + 128, G[5] + 192,
                                                      P[24], P[25], 64, MEDGE, zB, G[6]);
    bn_finalize<<<1, 64, 0, stream>>>(G[6], P[27], P[26], invEdges, 64, G[6] + 128, G[6] + 192);
    gemm_bnstats<1, 0, 2, 4><<<896, 256, 0, stream>>>(nullptr, nullptr, zB, G[6] + 128, G[6] + 192,
                                                      P[28], P[29], 64, MEDGE, zA, G[7]);
    bn_finalize<<<1, 64, 0, stream>>>(G[7], P[31], P[30], invEdges, 64, G[7] + 128, G[7] + 192);
    gemm_bnstats<2, 32, 1, 4><<<128, 256, 0, stream>>>(fts, nullptr, nullptr, nullptr, nullptr,
                                                       P[32], P[33], 32, NUMN, zsc, G[8]);
    bn_finalize<<<1, 64, 0, stream>>>(G[8], P[35], P[34], invNodes, 64, G[8] + 128, G[8] + 192);
    {
        const size_t n = (size_t)NUMN * 64;
        combine_edges<<<(int)((n + 255) / 256), 256, 0, stream>>>(
            zA, G[7] + 128, G[7] + 192, zsc, G[8] + 128, G[8] + 192, fts, 64);
    }

    // ---- pool + FC head ----
    head_kernel<<<BATCH, 128, 0, stream>>>(fts, P[36], P[38], P[37], P[39], out);
}
